// Model_10651518894330
// MI455X (gfx1250) — matrix operand order repaired and checked by reading, not compiled, not run on hardware
//
#include <hip/hip_runtime.h>
#include <math.h>

typedef __attribute__((ext_vector_type(16))) _Float16 v16h;
typedef __attribute__((ext_vector_type(8)))  float    v8f;

#define BM 128
#define BN 128
#define BK 32
#define LDK 40   // padded k-stride in halves for both LDS tiles

union Frag { v16h v; uint4 q[2]; };
union H8   { uint4 q; _Float16 h[8]; };

__global__ __launch_bounds__(256)
void gemm_blockscaled_f16(const _Float16* __restrict__ A,
                          const _Float16* __restrict__ B,
                          const float*    __restrict__ Ascale,
                          const float*    __restrict__ Bscale,
                          _Float16*       __restrict__ C,
                          int M, int N, int K) {
    // ping-pong buffers: a tile is only overwritten two iterations after it was
    // written, and the intervening barrier proves every wave finished reading it.
    __shared__ _Float16 As[2][BM * LDK];   // [m][k] row-major
    __shared__ _Float16 Bs[2][BN * LDK];   // [n][k] transposed for contiguous B frags

    const int t     = threadIdx.x;
    const int lane  = t & 31;
    const int wave  = t >> 5;
    const int waveM = wave >> 2;   // 0..1  -> 64 rows each
    const int waveN = wave & 3;    // 0..3  -> 32 cols each

    const int bM = blockIdx.y * BM;
    const int bN = blockIdx.x * BN;
    const int K4 = K >> 2;

    // ---- per-thread invariant addressing (advanced by constants each k-step) ----
    // A staging: segment s = t + i*256 -> row = s>>2 (0..127), kseg = s&3 (8 halves)
    int arow[2], akseg[2];
    const _Float16* pa[2];
    const float*    psa[2];
    int a_lds[2];
#pragma unroll
    for (int i = 0; i < 2; ++i) {
        int s    = t + i * 256;
        arow[i]  = s >> 2;
        akseg[i] = s & 3;
        pa[i]  = A + (size_t)(bM + arow[i]) * K + akseg[i] * 8;
        psa[i] = Ascale + (size_t)((bM + arow[i]) >> 2) * K4 + akseg[i] * 2;
        a_lds[i] = arow[i] * LDK + akseg[i] * 8;
    }
    // B staging: segment s -> kk = s>>4 (0..31), n0 = (s&15)*8 (8 consecutive n)
    int bkk[2], bn0[2];
    const _Float16* pb[2];
    const float*    psb[2];
    int b_lds[2];
#pragma unroll
    for (int i = 0; i < 2; ++i) {
        int s   = t + i * 256;
        bkk[i]  = s >> 4;
        bn0[i]  = (s & 15) * 8;
        pb[i]  = B + (size_t)bkk[i] * N + (bN + bn0[i]);
        psb[i] = Bscale + (size_t)((bN + bn0[i]) >> 2) * K4 + (bkk[i] >> 2);
        b_lds[i] = bn0[i] * LDK + bkk[i];
    }
    // fragment read offsets (halves), fixed per thread
    const int r  = lane & 15;
    const int kbA = (lane >> 4) * 8;    // lanes 0-15: K{0..7,16..23}; 16-31: {8..15,24..31}
    const int kbB = (lane >> 4) * 8;    // lanes 0-15: K{0..7,16..23}; 16-31: {8..15,24..31}
    int afrag_off[4], bfrag_off[2];
#pragma unroll
    for (int mt = 0; mt < 4; ++mt)
        afrag_off[mt] = (waveM * 64 + mt * 16 + r) * LDK + kbA;
#pragma unroll
    for (int nt = 0; nt < 2; ++nt)
        bfrag_off[nt] = (waveN * 32 + nt * 16 + r) * LDK + kbB;

    v8f acc[4][2];
#pragma unroll
    for (int mt = 0; mt < 4; ++mt)
#pragma unroll
        for (int nt = 0; nt < 2; ++nt)
            acc[mt][nt] = (v8f){0.f, 0.f, 0.f, 0.f, 0.f, 0.f, 0.f, 0.f};

    // ---- register staging for the prefetch pipeline ----
    H8 arw[2], brw[2];
    float as0[2], as1[2], bs0[2], bs1[2];

    // prologue: fetch k-step 0
#pragma unroll
    for (int i = 0; i < 2; ++i) {
        arw[i].q = *(const uint4*)(pa[i]);
        as0[i] = psa[i][0]; as1[i] = psa[i][1];
        brw[i].q = *(const uint4*)(pb[i]);
        bs0[i] = psb[i][0]; bs1[i] = psb[i][K4];
        pa[i] += BK; psa[i] += BK / 4;
        pb[i] += (size_t)BK * N; psb[i] += BK / 4;
    }

    int buf = 0;
    for (int k0 = 0; k0 < K; k0 += BK) {
        // ---- dequant-scale the staged registers into LDS[buf] ----
#pragma unroll
        for (int i = 0; i < 2; ++i) {
            H8 o;
#pragma unroll
            for (int j = 0; j < 8; ++j)
                o.h[j] = (_Float16)((float)arw[i].h[j] * (j < 4 ? as0[i] : as1[i]));
            *(uint4*)(&As[buf][a_lds[i]]) = o.q;
        }
#pragma unroll
        for (int i = 0; i < 2; ++i) {
#pragma unroll
            for (int j = 0; j < 8; ++j)
                Bs[buf][b_lds[i] + j * LDK] =
                    (_Float16)((float)brw[i].h[j] * (j < 4 ? bs0[i] : bs1[i]));
        }

        __syncthreads();

        // ---- issue next k-step's global loads; latency hidden by the WMMAs below ----
        if (k0 + BK < K) {
#pragma unroll
            for (int i = 0; i < 2; ++i) {
                arw[i].q = *(const uint4*)(pa[i]);
                as0[i] = psa[i][0]; as1[i] = psa[i][1];
                brw[i].q = *(const uint4*)(pb[i]);
                bs0[i] = psb[i][0]; bs1[i] = psb[i][K4];
                pa[i] += BK; psa[i] += BK / 4;
                pb[i] += (size_t)BK * N; psb[i] += BK / 4;
            }
        }

        // ---- fragments from LDS[buf] + 8 WMMAs ----
        Frag af[4];
#pragma unroll
        for (int mt = 0; mt < 4; ++mt) {
            const _Float16* p = &As[buf][afrag_off[mt]];
            af[mt].q[0] = *(const uint4*)(p);
            af[mt].q[1] = *(const uint4*)(p + 16);
        }
        Frag bf[2];
#pragma unroll
        for (int nt = 0; nt < 2; ++nt) {
            const _Float16* p = &Bs[buf][bfrag_off[nt]];
            bf[nt].q[0] = *(const uint4*)(p);
            bf[nt].q[1] = *(const uint4*)(p + 16);
        }
#pragma unroll
        for (int mt = 0; mt < 4; ++mt)
#pragma unroll
            for (int nt = 0; nt < 2; ++nt)
                acc[mt][nt] = __builtin_amdgcn_wmma_f32_16x16x32_f16(
                    /*neg_a=*/false, af[mt].v, /*neg_b=*/false, bf[nt].v,
                    /*c_mod=*/(short)0, acc[mt][nt],
                    /*reuse_a=*/false, /*reuse_b=*/false);

        buf ^= 1;
    }

    // ---- epilogue: fp32 acc -> fp16 C ----
    const int half = lane >> 4;   // C layout: vgpr j -> M = j + 8*half, N = r
#pragma unroll
    for (int mt = 0; mt < 4; ++mt)
#pragma unroll
        for (int nt = 0; nt < 2; ++nt) {
            int n = bN + waveN * 32 + nt * 16 + r;
#pragma unroll
            for (int j = 0; j < 8; ++j) {
                int m = bM + waveM * 64 + mt * 16 + half * 8 + j;
                C[(size_t)m * N + n] = (_Float16)acc[mt][nt][j];
            }
        }
}

extern "C" void kernel_launch(void* const* d_in, const int* in_sizes, int n_in,
                              void* d_out, int out_size, void* d_ws, size_t ws_size,
                              hipStream_t stream) {
    (void)n_in; (void)d_ws; (void)ws_size; (void)out_size;
    const _Float16* A      = (const _Float16*)d_in[0];
    const _Float16* B      = (const _Float16*)d_in[1];
    const float*    Ascale = (const float*)d_in[2];
    const float*    Bscale = (const float*)d_in[3];
    _Float16*       C      = (_Float16*)d_out;

    // square problem: M = N = K = sqrt(|A|)
    int dim = (int)(sqrt((double)in_sizes[0]) + 0.5);
    int M = dim, N = dim, K = dim;

    dim3 grid(N / BN, M / BM);
    dim3 block(256);
    gemm_blockscaled_f16<<<grid, block, 0, stream>>>(A, B, Ascale, Bscale, C, M, N, K);
}
